// TAttention_45603962749198
// MI455X (gfx1250) — matrix operand order repaired and checked by reading, not compiled, not run on hardware
//
#include <hip/hip_runtime.h>

// ---------------------------------------------------------------------------
// CDNA5 (gfx1250, wave32) bf16-WMMA self-attention with TDM-fed K/V tiles.
//   d_out = [ out (B,S,Dm) fp32 | logits_raw (B,H,S,S) fp32 ]
//   ws    = Q | K | V | attn_out, each (B*S, Dm) bf16  (~50.3 MB total)
// ---------------------------------------------------------------------------

typedef __attribute__((ext_vector_type(16))) __bf16        v16bf;
typedef __attribute__((ext_vector_type(8)))  float         v8f;
typedef __attribute__((ext_vector_type(2)))  __bf16        bf16x2;
typedef __attribute__((ext_vector_type(4)))  unsigned int  u32x4;
typedef __attribute__((ext_vector_type(8)))  unsigned int  u32x8;

#define S_LEN  2048
#define DMODEL 768
#define NHEAD  12
#define HDIM   64
#define BATCH  4
#define MROWS  (BATCH * S_LEN)

// A-matrix (16-bit, 16x32) K packing: lanes 0-15 hold K=0..7,16..23;
// lanes 16-31 hold K=8..15,24..31. VGPR v packs (k, k+1).
__device__ __forceinline__ int kpatA(int v, int hi) {
    return ((v < 4) ? 2 * v : 2 * v + 8) + 8 * hi;
}
// B-matrix (16-bit, 32x16) K packing: lanes 0-15 hold K=0..7,16..23;
// lanes 16-31 hold K=8..15,24..31. VGPR v packs (k, k+1).
__device__ __forceinline__ int kpatB(int v, int hi) {
    return ((v < 4) ? 2 * v : 2 * v + 8) + 8 * hi;
}

__device__ __forceinline__ v8f wmma_bf16(v16bf a, v16bf b, v8f c) {
    // (neg_a, A, neg_b, B, c_mod, C, reuse_a, reuse_b)
    return __builtin_amdgcn_wmma_f32_16x16x32_bf16(false, a, false, b,
                                                   (short)0, c, false, false);
}

// ---------------------------------------------------------------------------
// TDM: DMA one 32x64 bf16 tile (row stride DMODEL elems) from global into LDS,
// padding each 128B row with 16B so the LDS row stride is 144B (bank-friendly).
// 2D descriptor: groups 2/3 omitted (VADDR2/3 = NULL).  All operands uniform.
// ---------------------------------------------------------------------------
__device__ __forceinline__ void tdm_load_tile32x64(unsigned int lds_off,
                                                   const __bf16* gsrc) {
    const unsigned long long ga = (unsigned long long)(const void*)gsrc;
    u32x4 g0;
    g0[0] = 1u;                                        // count=1, user D#
    g0[1] = lds_off;                                   // lds_addr
    g0[2] = (unsigned int)ga;                          // global_addr[31:0]
    g0[3] = ((unsigned int)(ga >> 32) & 0x01FFFFFFu)   // global_addr[56:32]
            | 0x80000000u;                             // type=2 ("image")
    u32x8 g1;
    g1[0] = (1u << 16)        // data_size = 2 bytes
          | (1u << 20)        // pad_enable
          | (4u << 22)        // pad_interval: 32 dwords (=128B row)
          | (3u << 25);       // pad_amount: 4 dwords (=16B)
    g1[1] = (unsigned int)DMODEL << 16;   // tensor_dim0 = 768 (lo16)
    g1[2] = 8192u << 16;                  // tensor_dim0 hi | tensor_dim1 lo16
    g1[3] = 64u << 16;                    // tensor_dim1 hi | tile_dim0 = 64
    g1[4] = 32u;                          // tile_dim1 = 32, tile_dim2 = 0
    g1[5] = (unsigned int)DMODEL;         // tensor_dim0_stride = 768
    g1[6] = 0u;
    g1[7] = 0u;
    asm volatile("tensor_load_to_lds %0, %1" :: "s"(g0), "s"(g1) : "memory");
}

// ---------------------------------------------------------------------------
// GEMM: Y[M,N] = A[M,K] @ W[K,N] * scale.  A fp32 or bf16, W fp32 (converted
// to bf16 while staging, so no DMA here).  128x128 tile, BK=32, 256 threads.
// ---------------------------------------------------------------------------
template <typename TIN>
__global__ __launch_bounds__(256)
void gemm_bf16_kernel(const TIN* __restrict__ A, const float* __restrict__ W,
                      __bf16* __restrict__ Ybf, float* __restrict__ Yf,
                      int M, int N, int K, float scale) {
    __shared__ __bf16 As[128][34];   // [m][k], row stride 68B (odd dwords)
    __shared__ __bf16 Bs[128][34];   // transposed: [n][k]

    const int tid  = threadIdx.x;
    const int lane = tid & 31;
    const int wave = tid >> 5;
    const int hi   = lane >> 4;
    const int ln   = lane & 15;
    const int wm   = (wave & 1) * 64;
    const int wn   = (wave >> 1) * 32;
    const int m0   = blockIdx.x * 128;
    const int n0   = blockIdx.y * 128;

    v8f acc[4][2] = {};

    for (int k0 = 0; k0 < K; k0 += 32) {
        for (int i = tid; i < 128 * 32; i += 256) {
            int r = i >> 5, c = i & 31;
            As[r][c] = (__bf16)(float)A[(size_t)(m0 + r) * K + (k0 + c)];
        }
        for (int i = tid; i < 32 * 128; i += 256) {
            int r = i >> 7, c = i & 127;
            Bs[c][r] = (__bf16)W[(size_t)(k0 + r) * N + (n0 + c)];
        }
        __syncthreads();

        v16bf bfr[2];
#pragma unroll
        for (int ns = 0; ns < 2; ++ns) {
            const int col = wn + ns * 16 + ln;
#pragma unroll
            for (int v = 0; v < 8; ++v) {
                bf16x2 p = *(const bf16x2*)&Bs[col][kpatB(v, hi)];
                bfr[ns][2 * v] = p.x;
                bfr[ns][2 * v + 1] = p.y;
            }
        }
#pragma unroll
        for (int ms = 0; ms < 4; ++ms) {
            const int row = wm + ms * 16 + ln;
            v16bf af;
#pragma unroll
            for (int v = 0; v < 8; ++v) {
                bf16x2 p = *(const bf16x2*)&As[row][kpatA(v, hi)];
                af[2 * v] = p.x;
                af[2 * v + 1] = p.y;
            }
            acc[ms][0] = wmma_bf16(af, bfr[0], acc[ms][0]);
            acc[ms][1] = wmma_bf16(af, bfr[1], acc[ms][1]);
        }
        __syncthreads();
    }

#pragma unroll
    for (int ms = 0; ms < 4; ++ms)
#pragma unroll
        for (int ns = 0; ns < 2; ++ns)
#pragma unroll
            for (int r = 0; r < 8; ++r) {
                const int mg = m0 + wm + ms * 16 + r + 8 * hi;
                const int ng = n0 + wn + ns * 16 + ln;
                const float v = acc[ms][ns][r] * scale;
                if (Ybf) Ybf[(size_t)mg * N + ng] = (__bf16)v;
                if (Yf)  Yf[(size_t)mg * N + ng]  = v;
            }
}

// ---------------------------------------------------------------------------
// Flash-style attention per (b,h).  8 waves x 16 query rows.  K/V tiles are
// double-buffered in LDS and filled by the Tensor Data Mover: wave 0 issues
// the DMA for tile i+1 before the block computes on tile i, then drains
// TENSORcnt and the workgroup barrier publishes the data.
// ---------------------------------------------------------------------------
__global__ __launch_bounds__(256)
void attn_kernel(const __bf16* __restrict__ Q, const __bf16* __restrict__ Kb,
                 const __bf16* __restrict__ Vb, const float* __restrict__ qmask,
                 float* __restrict__ logits, __bf16* __restrict__ Ao) {
    __shared__ __attribute__((aligned(16))) __bf16 Kt[2][32][72]; // 144B rows
    __shared__ __attribute__((aligned(16))) __bf16 Vt[2][32][72];
    __shared__ __bf16 Pb[8][16][34];                              // per-wave P

    const int tid  = threadIdx.x;
    const int lane = tid & 31;
    const int wave = tid >> 5;
    const int hi   = lane >> 4;
    const int ln   = lane & 15;
    const int bh   = blockIdx.y;
    const int b    = bh / NHEAD;
    const int h    = bh - b * NHEAD;
    const int q0   = blockIdx.x * 128 + wave * 16;

    const __bf16* Kbase = Kb + (size_t)(b * S_LEN) * DMODEL + h * HDIM;
    const __bf16* Vbase = Vb + (size_t)(b * S_LEN) * DMODEL + h * HDIM;

    // Preload Q A-fragments (two 32-wide d chunks) for this wave's 16 rows.
    const size_t rowQ = (size_t)(b * S_LEN + q0 + ln) * DMODEL + h * HDIM;
    v16bf aq[2];
#pragma unroll
    for (int c = 0; c < 2; ++c)
#pragma unroll
        for (int v = 0; v < 8; ++v) {
            bf16x2 p = *(const bf16x2*)(Q + rowQ + 32 * c + kpatA(v, hi));
            aq[c][2 * v] = p.x;
            aq[c][2 * v + 1] = p.y;
        }

    v8f o[4] = {};
    float rmax[8], rsum[8], qm[8];
#pragma unroll
    for (int r = 0; r < 8; ++r) {
        rmax[r] = -3.0e38f;
        rsum[r] = 0.0f;
        qm[r]   = qmask[b * S_LEN + q0 + r + 8 * hi];
    }

    // Prologue: DMA tile 0 into buffer 0.
    if (wave == 0) {
        tdm_load_tile32x64((unsigned int)(size_t)&Kt[0][0][0], Kbase);
        tdm_load_tile32x64((unsigned int)(size_t)&Vt[0][0][0], Vbase);
        __builtin_amdgcn_s_wait_tensorcnt(0);
    }
    __syncthreads();

    for (int kt = 0; kt < S_LEN; kt += 32) {
        const int cur = (kt >> 5) & 1;
        // Kick off DMA for the next tile into the other buffer (overlapped).
        if (wave == 0 && (kt + 32) < S_LEN) {
            tdm_load_tile32x64((unsigned int)(size_t)&Kt[cur ^ 1][0][0],
                               Kbase + (size_t)(kt + 32) * DMODEL);
            tdm_load_tile32x64((unsigned int)(size_t)&Vt[cur ^ 1][0][0],
                               Vbase + (size_t)(kt + 32) * DMODEL);
        }

        // Logits tile: S[16q x 32k] = Q(16x64) . K^T(64x32); 4 WMMAs.
        v8f c0 = {}, c1 = {};
#pragma unroll
        for (int ch = 0; ch < 2; ++ch) {
            v16bf bk0, bk1;
#pragma unroll
            for (int v = 0; v < 8; ++v) {
                bf16x2 p0 = *(const bf16x2*)&Kt[cur][ln][32 * ch + kpatB(v, hi)];
                bf16x2 p1 = *(const bf16x2*)&Kt[cur][16 + ln][32 * ch + kpatB(v, hi)];
                bk0[2 * v] = p0.x; bk0[2 * v + 1] = p0.y;
                bk1[2 * v] = p1.x; bk1[2 * v + 1] = p1.y;
            }
            c0 = wmma_bf16(aq[ch], bk0, c0);
            c1 = wmma_bf16(aq[ch], bk1, c1);
        }

        // Raw logits out + mask + online softmax (rows live in 16-lane groups).
        const float km0 = qmask[b * S_LEN + kt + ln];
        const float km1 = qmask[b * S_LEN + kt + 16 + ln];
#pragma unroll
        for (int r = 0; r < 8; ++r) {
            const float v0 = c0[r], v1 = c1[r];
            const size_t off =
                ((size_t)bh * S_LEN + (q0 + r + 8 * hi)) * S_LEN + kt + ln;
            logits[off]      = v0;        // logits_raw (pre-mask)
            logits[off + 16] = v1;

            const float l0 = v0 - 100000.0f * (1.0f - qm[r] * km0);
            const float l1 = v1 - 100000.0f * (1.0f - qm[r] * km1);
            float tmax = fmaxf(l0, l1);
#pragma unroll
            for (int m = 8; m >= 1; m >>= 1)
                tmax = fmaxf(tmax, __shfl_xor(tmax, m, 16));
            const float nm = fmaxf(rmax[r], tmax);
            const float al = __expf(rmax[r] - nm);
            const float p0 = __expf(l0 - nm);
            const float p1 = __expf(l1 - nm);
            float ps = p0 + p1;
#pragma unroll
            for (int m = 8; m >= 1; m >>= 1)
                ps += __shfl_xor(ps, m, 16);
            rsum[r] = rsum[r] * al + ps;
            rmax[r] = nm;
#pragma unroll
            for (int t = 0; t < 4; ++t) o[t][r] *= al;
            // C-layout -> LDS so we can reload P in A-layout for P.V
            Pb[wave][r + 8 * hi][ln]      = (__bf16)p0;
            Pb[wave][r + 8 * hi][16 + ln] = (__bf16)p1;
        }
        asm volatile("s_wait_dscnt 0" ::: "memory");  // P visible within wave

        v16bf ap;
#pragma unroll
        for (int v = 0; v < 8; ++v) {
            bf16x2 p = *(const bf16x2*)&Pb[wave][ln][kpatA(v, hi)];
            ap[2 * v] = p.x;
            ap[2 * v + 1] = p.y;
        }
        // O += P(16x32) . V(32x64); 4 WMMAs (one per 16-wide d tile).
#pragma unroll
        for (int t = 0; t < 4; ++t) {
            v16bf bv;
#pragma unroll
            for (int v = 0; v < 8; ++v) {
                const int kk = kpatB(v, hi);
                bv[2 * v]     = Vt[cur][kk][t * 16 + ln];
                bv[2 * v + 1] = Vt[cur][kk + 1][t * 16 + ln];
            }
            o[t] = wmma_bf16(ap, bv, o[t]);
        }

        // Drain the DMA for tile i+1, then publish to all waves.
        if (wave == 0) __builtin_amdgcn_s_wait_tensorcnt(0);
        __syncthreads();
    }

    // Normalize and store attn output (bf16) for the Wo projection.
#pragma unroll
    for (int r = 0; r < 8; ++r) {
        const float inv = 1.0f / rsum[r];
        const int mg = b * S_LEN + q0 + r + 8 * hi;
#pragma unroll
        for (int t = 0; t < 4; ++t)
            Ao[(size_t)mg * DMODEL + h * HDIM + t * 16 + ln] =
                (__bf16)(o[t][r] * inv);
    }
}

// ---------------------------------------------------------------------------
extern "C" void kernel_launch(void* const* d_in, const int* in_sizes, int n_in,
                              void* d_out, int out_size, void* d_ws, size_t ws_size,
                              hipStream_t stream) {
    const float* x  = (const float*)d_in[0];
    const float* qm = (const float*)d_in[1];
    const float* Wq = (const float*)d_in[2];
    const float* Wk = (const float*)d_in[3];
    const float* Wv = (const float*)d_in[4];
    const float* Wo = (const float*)d_in[5];

    float* out    = (float*)d_out;
    float* logits = out + (size_t)BATCH * S_LEN * DMODEL;

    const size_t elems = (size_t)MROWS * DMODEL;        // 8192*768
    __bf16* Qb = (__bf16*)d_ws;
    __bf16* Kb = Qb + elems;
    __bf16* Vb = Kb + elems;
    __bf16* Ab = Vb + elems;                            // ~50.3 MB total

    const dim3 gg(MROWS / 128, DMODEL / 128);           // 64 x 6
    const dim3 gb(256);
    const float qscale = 0.125f;                        // 1/sqrt(HDIM)

    gemm_bf16_kernel<float><<<gg, gb, 0, stream>>>(x, Wq, Qb, nullptr,
                                                   MROWS, DMODEL, DMODEL, qscale);
    gemm_bf16_kernel<float><<<gg, gb, 0, stream>>>(x, Wk, Kb, nullptr,
                                                   MROWS, DMODEL, DMODEL, 1.0f);
    gemm_bf16_kernel<float><<<gg, gb, 0, stream>>>(x, Wv, Vb, nullptr,
                                                   MROWS, DMODEL, DMODEL, 1.0f);

    attn_kernel<<<dim3(S_LEN / 128, BATCH * NHEAD), gb, 0, stream>>>(
        Qb, Kb, Vb, qm, logits, Ab);

    gemm_bf16_kernel<__bf16><<<gg, gb, 0, stream>>>(Ab, Wo, nullptr, out,
                                                    MROWS, DMODEL, DMODEL, 1.0f);
}
